// RippleBlock_39539468927752
// MI455X (gfx1250) — matrix operand order repaired and checked by reading, not compiled, not run on hardware
//
#include <hip/hip_runtime.h>
#include <cstdint>

// ---------------------------------------------------------------------------
// Problem constants (from the reference): B=2, L=2048, C=1024, H=16, D=64
// KC = int(2048^0.421875) = 24, LH = int(2048^0.75) = 304, LT = int(2048^0.5625) = 72
// ---------------------------------------------------------------------------
#define B_   2
#define L_   2048
#define C_   1024
#define H_   16
#define D_   64
#define KC_  24
#define LH_  304
#define LT_  72
#define EPS_ 1e-6f

typedef __bf16          v16bf   __attribute__((ext_vector_type(16)));
typedef float           v8f     __attribute__((ext_vector_type(8)));
typedef unsigned short  ushort8 __attribute__((ext_vector_type(8)));

union BFrag { v16bf v; ushort8 h[2]; };

// float -> bf16 with round-to-nearest-even
static __device__ inline unsigned short f2bf(float f) {
  unsigned u = __builtin_bit_cast(unsigned, f);
  u += 0x7fffu + ((u >> 16) & 1u);
  return (unsigned short)(u >> 16);
}

// ---------------------------------------------------------------------------
// GEMM: C[M,N] = act(A[M,K] @ W[K,N] + bias) + resid
// bf16 WMMA (v_wmma_f32_16x16x32_bf16), fp32 accumulate.
// Block = 256 threads (8 waves). Block tile 128x128, BK=32, software-pipelined
// global->reg staging. Wave (wm 0..3, wn 0..1) computes a 32x64 sub-tile =
// 2x4 WMMA accumulators -> 8 WMMAs per wave per K-step against 12 ds_load_b128
// (A frags reused 4x, B frags reused 2x).
// N,K must be multiples of 128/32 (true for 1024/4096/384); M bound-guarded.
// Fragment layouts follow CDNA5 ISA 7.12.2:
//  A 16x32 bf16: lane<16 holds row=lane, K {0..7,16..23}; lane>=16 K {8..15,24..31}
//  B 32x16 bf16: lane<16 holds K {0..7,16..23}, lane>=16 K {8..15,24..31}; col = lane%16
//  C/D: vgpr e -> row = e + (lane>=16 ? 8:0), col = lane%16
// ---------------------------------------------------------------------------
__global__ __launch_bounds__(256) void gemm_bf16_wmma(
    const float* __restrict__ A, const float* __restrict__ W,
    float* __restrict__ Cout, int M, int N, int K,
    const float* __restrict__ bias, int act, const float* __restrict__ resid)
{
  __shared__ __align__(16) unsigned short As[128 * 32];   // [row][k]
  __shared__ __align__(16) unsigned short Bs[128 * 32];   // [col][k] (transposed)

  const int tid  = threadIdx.x;
  const int m0   = blockIdx.y * 128, n0 = blockIdx.x * 128;
  const int lane = tid & 31, wave = tid >> 5;
  const int wm   = wave >> 1, wn = wave & 1;
  const int r    = lane & 15, half = lane >> 4;

  // loader indices
  const int ar = tid >> 1, ac = (tid & 1) * 16;   // A: 128 rows x 2 halves of 16
  const int wk = tid >> 3, wng = tid & 7;         // W: 32 k-rows x 8 groups of 16 cols

  ushort8 aReg0 = {}, aReg1 = {};
  unsigned short wReg[16];

  auto loadA = [&](int k0) {
    float4 f0 = make_float4(0.f,0.f,0.f,0.f), f1 = f0, f2 = f0, f3 = f0;
    if (m0 + ar < M) {
      const float* ap = A + (long long)(m0 + ar) * K + k0 + ac;
      f0 = *(const float4*)(ap);
      f1 = *(const float4*)(ap + 4);
      f2 = *(const float4*)(ap + 8);
      f3 = *(const float4*)(ap + 12);
      if (k0 + 32 < K) __builtin_prefetch(ap + 32, 0, 1);   // global_prefetch_b8
    }
    aReg0[0]=f2bf(f0.x); aReg0[1]=f2bf(f0.y); aReg0[2]=f2bf(f0.z); aReg0[3]=f2bf(f0.w);
    aReg0[4]=f2bf(f1.x); aReg0[5]=f2bf(f1.y); aReg0[6]=f2bf(f1.z); aReg0[7]=f2bf(f1.w);
    aReg1[0]=f2bf(f2.x); aReg1[1]=f2bf(f2.y); aReg1[2]=f2bf(f2.z); aReg1[3]=f2bf(f2.w);
    aReg1[4]=f2bf(f3.x); aReg1[5]=f2bf(f3.y); aReg1[6]=f2bf(f3.z); aReg1[7]=f2bf(f3.w);
  };
  auto loadW = [&](int k0) {
    const float* wp = W + (long long)(k0 + wk) * N + n0 + wng * 16;
    float4 f0 = *(const float4*)(wp);
    float4 f1 = *(const float4*)(wp + 4);
    float4 f2 = *(const float4*)(wp + 8);
    float4 f3 = *(const float4*)(wp + 12);
    wReg[ 0]=f2bf(f0.x); wReg[ 1]=f2bf(f0.y); wReg[ 2]=f2bf(f0.z); wReg[ 3]=f2bf(f0.w);
    wReg[ 4]=f2bf(f1.x); wReg[ 5]=f2bf(f1.y); wReg[ 6]=f2bf(f1.z); wReg[ 7]=f2bf(f1.w);
    wReg[ 8]=f2bf(f2.x); wReg[ 9]=f2bf(f2.y); wReg[10]=f2bf(f2.z); wReg[11]=f2bf(f2.w);
    wReg[12]=f2bf(f3.x); wReg[13]=f2bf(f3.y); wReg[14]=f2bf(f3.z); wReg[15]=f2bf(f3.w);
  };

  v8f acc[2][4] = {};

  loadA(0); loadW(0);
  for (int k0 = 0; k0 < K; k0 += 32) {
    // commit staged tile to LDS
    *(ushort8*)&As[ar * 32 + ac]     = aReg0;
    *(ushort8*)&As[ar * 32 + ac + 8] = aReg1;
#pragma unroll
    for (int i = 0; i < 16; ++i)
      Bs[(wng * 16 + i) * 32 + wk] = wReg[i];
    __syncthreads();

    // stage next tile while WMMAs run on this one
    if (k0 + 32 < K) { loadA(k0 + 32); loadW(k0 + 32); }

    BFrag afr[2], bfr[4];
#pragma unroll
    for (int mt = 0; mt < 2; ++mt) {
      const unsigned short* ab = &As[(wm * 32 + mt * 16 + r) * 32 + half * 8];
      afr[mt].h[0] = *(const ushort8*)(ab);        // K {0..7} / {8..15}
      afr[mt].h[1] = *(const ushort8*)(ab + 16);   // K {16..23} / {24..31}
    }
#pragma unroll
    for (int nt = 0; nt < 4; ++nt) {
      const unsigned short* bb = &Bs[(wn * 64 + nt * 16 + r) * 32 + half * 8];
      bfr[nt].h[0] = *(const ushort8*)(bb);
      bfr[nt].h[1] = *(const ushort8*)(bb + 16);
    }
#pragma unroll
    for (int mt = 0; mt < 2; ++mt)
#pragma unroll
      for (int nt = 0; nt < 4; ++nt)
        acc[mt][nt] = __builtin_amdgcn_wmma_f32_16x16x32_bf16(
            false, afr[mt].v, false, bfr[nt].v, (short)0, acc[mt][nt], false, false);
    __syncthreads();
  }

  // -- epilogue: bias / silu / residual
#pragma unroll
  for (int mt = 0; mt < 2; ++mt) {
#pragma unroll
    for (int e = 0; e < 8; ++e) {
      int row = m0 + wm * 32 + mt * 16 + e + half * 8;
      if (row >= M) continue;
#pragma unroll
      for (int nt = 0; nt < 4; ++nt) {
        int col = n0 + wn * 64 + nt * 16 + r;
        float v = acc[mt][nt][e];
        if (bias) v += bias[col];
        if (act == 1) v = v * (1.0f / (1.0f + __expf(-v)));   // silu
        long long off = (long long)row * N + col;
        if (resid) v += resid[off];
        Cout[off] = v;
      }
    }
  }
}

// ---------------------------------------------------------------------------
// Flash attention (no masking): O = softmax(Q K^T / sqrt(D)) V
// Q:(B,Lq,H*D) K,V:(B,Lk,H*D). Grid (ceil(Lq/64), H, B), 256 threads.
// 4 lanes per query (16 dims each); K/V chunks of 64 rows staged into LDS via
// GLOBAL_LOAD_ASYNC_TO_LDS_B128 (ASYNCcnt path, CDNA5 ISA)
// -- no VGPR round-trip, completion via s_wait_asynccnt + barrier.
// Flat pointers to LDS carry the wave-relative LDS offset in their low 32 bits
// (ISA §10.2), which is exactly the VDST operand the instruction wants.
// ---------------------------------------------------------------------------
__global__ __launch_bounds__(256) void attn_flash(
    const float* __restrict__ Q, const float* __restrict__ K,
    const float* __restrict__ V, float* __restrict__ O, int Lq, int Lk)
{
  __shared__ __align__(16) float Ks[64][68];
  __shared__ __align__(16) float Vs[64][68];
  const int tid = threadIdx.x;
  const int b = blockIdx.z, h = blockIdx.y;
  const int qi = tid >> 2, tq = tid & 3;
  const int qglob = blockIdx.x * 64 + qi;
  const float scale = 0.125f;   // D^-0.5, D=64

  float qf[16];
  {
    const long long qb = ((long long)(b * Lq + (qglob < Lq ? qglob : 0))) * C_ + h * D_ + tq * 16;
#pragma unroll
    for (int d = 0; d < 16; ++d) qf[d] = (qglob < Lq) ? Q[qb + d] : 0.0f;
  }
  float of[16];
#pragma unroll
  for (int d = 0; d < 16; ++d) of[d] = 0.0f;
  float mval = -1e30f, lsum = 0.0f;

  const int jr = tid >> 2, seg = tid & 3;
  const unsigned ldsK = (unsigned)(uintptr_t)&Ks[jr][seg * 16];
  const unsigned ldsV = (unsigned)(uintptr_t)&Vs[jr][seg * 16];

  for (int kc = 0; kc < Lk; kc += 64) {
    const int jglob = kc + jr;
    if (jglob < Lk) {
      const unsigned long long kg =
          (unsigned long long)(uintptr_t)(K + ((long long)(b * Lk + jglob)) * C_ + h * D_ + seg * 16);
      const unsigned long long vg =
          (unsigned long long)(uintptr_t)(V + ((long long)(b * Lk + jglob)) * C_ + h * D_ + seg * 16);
#pragma unroll
      for (int i = 0; i < 4; ++i) {
        asm volatile("global_load_async_to_lds_b128 %0, %1, off"
                     :: "v"(ldsK + (unsigned)(i * 16)), "v"(kg + (unsigned long long)(i * 16))
                     : "memory");
        asm volatile("global_load_async_to_lds_b128 %0, %1, off"
                     :: "v"(ldsV + (unsigned)(i * 16)), "v"(vg + (unsigned long long)(i * 16))
                     : "memory");
      }
    }
    asm volatile("s_wait_asynccnt 0x0" ::: "memory");   // this wave's copies landed
    __syncthreads();                                    // all waves' copies landed

    int jmax = Lk - kc; if (jmax > 64) jmax = 64;
    for (int j = 0; j < jmax; ++j) {
      float p = 0.0f;
#pragma unroll
      for (int d = 0; d < 16; ++d) p += qf[d] * Ks[j][tq * 16 + d];
      p += __shfl_xor(p, 1, 32);     // reduce over the 4 lanes of this query
      p += __shfl_xor(p, 2, 32);
      float s    = p * scale;
      float mnew = fmaxf(mval, s);
      float corr = __expf(mval - mnew);
      float w    = __expf(s - mnew);
      lsum = lsum * corr + w;
#pragma unroll
      for (int d = 0; d < 16; ++d) of[d] = of[d] * corr + w * Vs[j][tq * 16 + d];
      mval = mnew;
    }
    __syncthreads();
  }

  if (qglob < Lq) {
    float inv = 1.0f / lsum;
    float* op = O + ((long long)(b * Lq + qglob)) * C_ + h * D_ + tq * 16;
#pragma unroll
    for (int d = 0; d < 16; ++d) op[d] = of[d] * inv;
  }
}

// ---------------------------------------------------------------------------
// Per-head RMS norm (+ optional depth-RoPE with depth=1). One wave per head.
// Lane owns dims {lane, lane+32} of the 64-dim head.
// ---------------------------------------------------------------------------
__global__ __launch_bounds__(512) void headrms_rope(
    float* __restrict__ X, const float* __restrict__ wgt, int Lq, int do_rope)
{
  const int tid = threadIdx.x;
  const int head = tid >> 5, lane = tid & 31;
  const int l = blockIdx.x, b = blockIdx.y;
  float* p = X + ((long long)(b * Lq + l)) * C_ + head * D_;
  float v1 = p[lane], v2 = p[lane + 32];
  float ss = v1 * v1 + v2 * v2;
#pragma unroll
  for (int off = 16; off; off >>= 1) ss += __shfl_xor(ss, off, 32);
  float scale = rsqrtf(ss * (1.0f / 64.0f) + EPS_);
  float a1 = v1 * scale * wgt[lane];
  float a2 = v2 * scale * wgt[lane + 32];
  if (do_rope) {
    float f = __powf(10000.0f, -(float)lane * (1.0f / 32.0f));
    float c = __cosf(f), s = __sinf(f);           // depth = 1.0
    float o1 = a1 * c - a2 * s;
    float o2 = a1 * s + a2 * c;
    a1 = o1; a2 = o2;
  }
  p[lane] = a1; p[lane + 32] = a2;
}

// ---------------------------------------------------------------------------
// out[b,l,:] = (resid ? resid[b,l,:] : 0) + rms(interp1d(src, Lout)[b,l,:], wgt)
// Lin==Lout degenerates to a plain rms-add. Grid (Lout, B), 256 threads/row.
// ---------------------------------------------------------------------------
__global__ __launch_bounds__(256) void interp_rms_add(
    const float* __restrict__ src, int Lin, const float* __restrict__ resid,
    const float* __restrict__ wgt, float* __restrict__ out, int Lout)
{
  __shared__ float y[C_];
  __shared__ float red[256];
  const int tid = threadIdx.x;
  const int l = blockIdx.x, b = blockIdx.y;
  float pos = ((float)l + 0.5f) * ((float)Lin / (float)Lout) - 0.5f;
  pos = fminf(fmaxf(pos, 0.0f), (float)(Lin - 1));
  int lo0 = (int)pos;
  int hi0 = lo0 + 1; if (hi0 > Lin - 1) hi0 = Lin - 1;
  float w = pos - (float)lo0;
  const float* s0 = src + ((long long)(b * Lin + lo0)) * C_;
  const float* s1 = src + ((long long)(b * Lin + hi0)) * C_;
  float acc = 0.0f;
  for (int c = tid; c < C_; c += 256) {
    float v = s0[c] * (1.0f - w) + s1[c] * w;
    y[c] = v;
    acc += v * v;
  }
  red[tid] = acc;
  __syncthreads();
  for (int s = 128; s; s >>= 1) {
    if (tid < s) red[tid] += red[tid + s];
    __syncthreads();
  }
  float scale = rsqrtf(red[0] * (1.0f / (float)C_) + EPS_);
  float* o = out + ((long long)(b * Lout + l)) * C_;
  const float* rp = resid ? resid + ((long long)(b * Lout + l)) * C_ : nullptr;
  for (int c = tid; c < C_; c += 256) {
    float v = y[c] * scale * wgt[c];
    if (rp) v += rp[c];
    o[c] = v;
  }
}

// Plain linear interpolation along L: out (B,Lout,C). Grid (Lout, B).
__global__ __launch_bounds__(256) void interp_down(
    const float* __restrict__ src, int Lin, float* __restrict__ out, int Lout)
{
  const int tid = threadIdx.x;
  const int l = blockIdx.x, b = blockIdx.y;
  float pos = ((float)l + 0.5f) * ((float)Lin / (float)Lout) - 0.5f;
  pos = fminf(fmaxf(pos, 0.0f), (float)(Lin - 1));
  int lo0 = (int)pos;
  int hi0 = lo0 + 1; if (hi0 > Lin - 1) hi0 = Lin - 1;
  float w = pos - (float)lo0;
  const float* s0 = src + ((long long)(b * Lin + lo0)) * C_;
  const float* s1 = src + ((long long)(b * Lin + hi0)) * C_;
  float* o = out + ((long long)(b * Lout + l)) * C_;
  for (int c = tid; c < C_; c += 256)
    o[c] = s0[c] * (1.0f - w) + s1[c] * w;
}

// ---------------------------------------------------------------------------
// Adaptive conv: O[b,l,h,d] = sum_k softmax(gen[b,l,h,:])[k] * H[b, clip(l+k-12), h, d]
// One wave per head; lane owns dims {2*lane, 2*lane+1}. Grid (L, B), 512 thr.
// ---------------------------------------------------------------------------
__global__ __launch_bounds__(512) void adaptive_conv_k(
    const float* __restrict__ Hx, const float* __restrict__ gen, float* __restrict__ O)
{
  const int tid = threadIdx.x;
  const int head = tid >> 5, lane = tid & 31;
  const int l = blockIdx.x, b = blockIdx.y;
  const long long grow = ((long long)(b * L_ + l)) * (H_ * KC_) + head * KC_;
  float g = (lane < KC_) ? gen[grow + lane] : -1e30f;
  float mx = g;
#pragma unroll
  for (int off = 16; off; off >>= 1) mx = fmaxf(mx, __shfl_xor(mx, off, 32));
  float e = (lane < KC_) ? __expf(g - mx) : 0.0f;
  float se = e;
#pragma unroll
  for (int off = 16; off; off >>= 1) se += __shfl_xor(se, off, 32);
  float wt = e / se;
  float ax = 0.0f, ay = 0.0f;
#pragma unroll
  for (int k = 0; k < KC_; ++k) {
    float wk = __shfl(wt, k, 32);
    int idx = l + k - KC_ / 2;
    idx = idx < 0 ? 0 : (idx > L_ - 1 ? L_ - 1 : idx);
    const float2 hv = *(const float2*)(Hx + ((long long)(b * L_ + idx)) * C_ + head * D_ + lane * 2);
    ax += wk * hv.x; ay += wk * hv.y;
  }
  float2 res; res.x = ax; res.y = ay;
  *(float2*)(O + ((long long)(b * L_ + l)) * C_ + head * D_ + lane * 2) = res;
}

__global__ __launch_bounds__(256) void addk(
    float* __restrict__ a, const float* __restrict__ bsrc, long long n)
{
  long long i = (long long)blockIdx.x * 256 + threadIdx.x;
  if (i < n) a[i] += bsrc[i];
}

// ---------------------------------------------------------------------------
// Orchestration
// ---------------------------------------------------------------------------
extern "C" void kernel_launch(void* const* d_in, const int* in_sizes, int n_in,
                              void* d_out, int out_size, void* d_ws, size_t ws_size,
                              hipStream_t stream)
{
  (void)in_sizes; (void)n_in; (void)out_size; (void)ws_size;

  const float* x     = (const float*)d_in[0];
  const float* hist0 = (const float*)d_in[1];
  const float* hist1 = (const float*)d_in[2];
  const float* cq    = (const float*)d_in[3];
  const float* ck    = (const float*)d_in[4];
  const float* cv    = (const float*)d_in[5];
  const float* co    = (const float*)d_in[6];
  const float* cqn   = (const float*)d_in[7];
  const float* ckn   = (const float*)d_in[8];
  const float* con   = (const float*)d_in[9];
  const float* tq    = (const float*)d_in[10];
  const float* tk    = (const float*)d_in[11];
  const float* tv    = (const float*)d_in[12];
  const float* to    = (const float*)d_in[13];
  const float* wgen  = (const float*)d_in[14];
  const float* wco   = (const float*)d_in[15];
  const float* lq    = (const float*)d_in[16];
  const float* lk    = (const float*)d_in[17];
  const float* lv    = (const float*)d_in[18];
  const float* lo    = (const float*)d_in[19];
  const float* ntele = (const float*)d_in[20];
  const float* nconv = (const float*)d_in[21];
  const float* nlow  = (const float*)d_in[22];
  const float* w1    = (const float*)d_in[23];
  const float* b1    = (const float*)d_in[24];
  const float* w2    = (const float*)d_in[25];
  const float* b2    = (const float*)d_in[26];
  const float* hn    = (const float*)d_in[27];

  float* out = (float*)d_out;
  float* ws  = (float*)d_ws;

  const long long S_BIG  = (long long)B_ * L_ * C_;       // 4,194,304
  const long long S_HIST = (long long)B_ * LH_ * C_;      //   622,592
  const long long S_U    = (long long)B_ * L_ * 4 * C_;   // 16,777,216

  float* X1 = ws;                  // x after history injection (later x2 in-place)
  float* Hh = ws + 1 * S_BIG;      // running h
  float* T0 = ws + 2 * S_BIG;      // projections
  float* T1 = ws + 3 * S_BIG;      // attention out
  float* T2 = ws + 4 * S_BIG;      // @Wo out / conv out
  float* U  = ws + 5 * S_BIG;      // MLP hidden
  float* q0 = ws + 5 * S_BIG + S_U;
  float* k0 = q0 + 1 * S_HIST;
  float* v0 = q0 + 2 * S_HIST;
  float* o0 = q0 + 3 * S_HIST;
  float* o0c= q0 + 4 * S_HIST;
  float* xk = q0 + 5 * S_HIST;
  float* gen= q0 + 6 * S_HIST;

  const dim3 blk(256);
  const int Mh = B_ * LH_;    // 608
  const int Mx = B_ * L_;     // 4096
  const int Mt = B_ * LT_;    // 144
  auto g128 = [](int M, int N) { return dim3((unsigned)((N + 127) / 128), (unsigned)((M + 127) / 128)); };

  // ---- Stage A: cross-attention over history, injected into x
  gemm_bf16_wmma<<<g128(Mh, C_), blk, 0, stream>>>(hist1, cq, q0, Mh, C_, C_, nullptr, 0, nullptr);
  headrms_rope<<<dim3(LH_, B_), 512, 0, stream>>>(q0, cqn, LH_, 1);
  gemm_bf16_wmma<<<g128(Mh, C_), blk, 0, stream>>>(hist0, ck, k0, Mh, C_, C_, nullptr, 0, nullptr);
  headrms_rope<<<dim3(LH_, B_), 512, 0, stream>>>(k0, ckn, LH_, 0);   // depth=0 rope is identity
  gemm_bf16_wmma<<<g128(Mh, C_), blk, 0, stream>>>(hist0, cv, v0, Mh, C_, C_, nullptr, 0, nullptr);
  attn_flash<<<dim3((LH_ + 63) / 64, H_, B_), blk, 0, stream>>>(q0, k0, v0, o0, LH_, LH_);
  gemm_bf16_wmma<<<g128(Mh, C_), blk, 0, stream>>>(o0, co, o0c, Mh, C_, C_, nullptr, 0, nullptr);
  interp_rms_add<<<dim3(L_, B_), blk, 0, stream>>>(o0c, LH_, x, con, X1, L_);     // x1

  // ---- Stage B: telescopic attention (kv length 72)
  interp_down<<<dim3(LT_, B_), blk, 0, stream>>>(X1, L_, xk, LT_);
  gemm_bf16_wmma<<<g128(Mx, C_), blk, 0, stream>>>(X1, tq, T0, Mx, C_, C_, nullptr, 0, nullptr);
  gemm_bf16_wmma<<<g128(Mt, C_), blk, 0, stream>>>(xk, tk, k0, Mt, C_, C_, nullptr, 0, nullptr);
  gemm_bf16_wmma<<<g128(Mt, C_), blk, 0, stream>>>(xk, tv, v0, Mt, C_, C_, nullptr, 0, nullptr);
  attn_flash<<<dim3((L_ + 63) / 64, H_, B_), blk, 0, stream>>>(T0, k0, v0, T1, L_, LT_);
  gemm_bf16_wmma<<<g128(Mx, C_), blk, 0, stream>>>(T1, to, T2, Mx, C_, C_, nullptr, 0, nullptr);
  interp_rms_add<<<dim3(L_, B_), blk, 0, stream>>>(T2, L_, X1, ntele, Hh, L_);    // h1

  // ---- Stage C: adaptive conv
  gemm_bf16_wmma<<<g128(Mx, H_ * KC_), blk, 0, stream>>>(Hh, wgen, gen, Mx, H_ * KC_, C_, nullptr, 0, nullptr);
  adaptive_conv_k<<<dim3(L_, B_), 512, 0, stream>>>(Hh, gen, T2);
  gemm_bf16_wmma<<<g128(Mx, C_), blk, 0, stream>>>(T2, wco, T0, Mx, C_, C_, nullptr, 0, nullptr);
  interp_rms_add<<<dim3(L_, B_), blk, 0, stream>>>(T0, L_, Hh, nconv, Hh, L_);    // h2

  // ---- Stage D: low attention (kv length 304)
  interp_down<<<dim3(LH_, B_), blk, 0, stream>>>(Hh, L_, xk, LH_);
  gemm_bf16_wmma<<<g128(Mx, C_), blk, 0, stream>>>(Hh, lq, T0, Mx, C_, C_, nullptr, 0, nullptr);
  gemm_bf16_wmma<<<g128(Mh, C_), blk, 0, stream>>>(xk, lk, k0, Mh, C_, C_, nullptr, 0, nullptr);
  gemm_bf16_wmma<<<g128(Mh, C_), blk, 0, stream>>>(xk, lv, v0, Mh, C_, C_, nullptr, 0, nullptr);
  attn_flash<<<dim3((L_ + 63) / 64, H_, B_), blk, 0, stream>>>(T0, k0, v0, T1, L_, LH_);
  gemm_bf16_wmma<<<g128(Mx, C_), blk, 0, stream>>>(T1, lo, T2, Mx, C_, C_, nullptr, 0, nullptr);
  interp_rms_add<<<dim3(L_, B_), blk, 0, stream>>>(T2, L_, Hh, nlow, Hh, L_);     // h3

  // ---- x2 = x1 + h  (in place into X1)
  addk<<<dim3((unsigned)((S_BIG + 255) / 256)), blk, 0, stream>>>(X1, Hh, S_BIG);

  // ---- MLP: out = x2 + silu(x2@w1 + b1)@w2 + b2
  gemm_bf16_wmma<<<g128(Mx, 4 * C_), blk, 0, stream>>>(X1, w1, U, Mx, 4 * C_, C_, b1, 1, nullptr);
  gemm_bf16_wmma<<<g128(Mx, C_), blk, 0, stream>>>(U, w2, out, Mx, C_, 4 * C_, b2, 0, X1);

  // ---- x_low = rms(interp1d(x, LH), hn) appended after x in d_out
  interp_rms_add<<<dim3(LH_, B_), blk, 0, stream>>>(out, L_, nullptr, hn, out + S_BIG, LH_);
}
